// LowerLinear_42984032698759
// MI455X (gfx1250) — compile-verified, matrix operand order checked by reading, not run on hardware
//
#include <hip/hip_runtime.h>
#include <hip/hip_bf16.h>

typedef __attribute__((ext_vector_type(16))) _Float16 v16h;
typedef __attribute__((ext_vector_type(8)))  _Float16 v8h;
typedef __attribute__((ext_vector_type(2)))  _Float16 v2h;
typedef __attribute__((ext_vector_type(8)))  float    v8f;
typedef __attribute__((__vector_size__(16))) int      v4i;

#define M_TOTAL 16384   // B * S = 8 * 2048
#define N_TOTAL 4096    // OUT
#define K_TOTAL 4096    // IN
#define GROUP   32      // quant block size along K
#define KBLOCKS (K_TOTAL / GROUP)               // 128 blocks per output row
#define NB_TOTAL (N_TOTAL * KBLOCKS)            // 524288 quant blocks

#define BM 256
#define BN 128
#define BK 32
#define BKP 40          // padded K stride in halves (80 B) vs LDS bank conflicts
#define NKT (K_TOTAL / BK)                      // 128 K-tiles

// CDNA5 async global->LDS path (guarded so the file compiles on any toolchain)
#if defined(__AMDGCN__) && __has_builtin(__builtin_amdgcn_global_load_async_to_lds_b128)
#define USE_ASYNC_LDS 1
#else
#define USE_ASYNC_LDS 0
#endif

#if USE_ASYNC_LDS
// param0: int4 AS(1)* (non-const), param1: int4 AS(3)*, then offset, cpol
__device__ __forceinline__ void async_copy_b128(const _Float16* g, _Float16* l) {
    __builtin_amdgcn_global_load_async_to_lds_b128(
        (__attribute__((address_space(1))) v4i*)(g),
        (__attribute__((address_space(3))) v4i*)(l), 0, 0);
}
#endif

__device__ __forceinline__ void wait_async_zero() {
#if __has_builtin(__builtin_amdgcn_s_wait_asynccnt)
    __builtin_amdgcn_s_wait_asynccnt(0);
#elif defined(__AMDGCN__)
    asm volatile("s_wait_asynccnt 0" ::: "memory");
#endif
}

// ---------------------------------------------------------------------------
// int4 pair -> packed f16 pair via exponent-bias trick:
//   bitcast(0x6400 | q) == 1024 + q exactly in fp16
//   (h - 1032) * s == (q - 8) * s   (q-8 exact, single rounding at the mul)
__device__ __forceinline__ unsigned dq_pair(int v, _Float16 sH) {
    unsigned t = (unsigned)v | ((unsigned)v << 12);          // nib0 @[3:0], nib1 @[19:16]
    unsigned u = (t & 0x000F000Fu) | 0x64006400u;
    v2h hv = __builtin_bit_cast(v2h, u);
    hv = (hv - (_Float16)1032.0f) * sH;                      // v_pk_add_f16 + v_pk_mul_f16
    return __builtin_bit_cast(unsigned, hv);
}

__device__ __forceinline__ _Float16 eff_scale(_Float16 s, unsigned char m) {
    float sf = (float)s;
    return (_Float16)(m ? sf : 0.5f * sf);
}

// ---------------------------------------------------------------------------
// Pre-pass 1: dequantize the whole weight matrix once (16.7M elements).
// Output layout == row-major [N][K] f16 (block nb covers row n=nb/128, k=32*(nb%128)).
__global__ __launch_bounds__(256)
void dequant_w_kernel(const int* __restrict__ wp,
                      const _Float16* __restrict__ wscale,
                      const unsigned char* __restrict__ wmask,
                      _Float16* __restrict__ wout)
{
    const int nb = blockIdx.x * 256 + threadIdx.x;           // 0 .. NB_TOTAL-1
    const int4* p = (const int4*)(wp + (size_t)nb * (GROUP / 2));
    int4 q[4];
    q[0] = p[0]; q[1] = p[1]; q[2] = p[2]; q[3] = p[3];

    const _Float16 sH = eff_scale(wscale[nb], wmask[nb]);

    uint4* o = (uint4*)(wout + (size_t)nb * GROUP);
    #pragma unroll
    for (int i = 0; i < 4; ++i) {
        uint4 r;
        r.x = dq_pair(((const int*)&q[i])[0], sH);
        r.y = dq_pair(((const int*)&q[i])[1], sH);
        r.z = dq_pair(((const int*)&q[i])[2], sH);
        r.w = dq_pair(((const int*)&q[i])[3], sH);
        o[i] = r;
    }
}

// ---------------------------------------------------------------------------
// Pre-pass 2: convert x fp32 -> f16 once (67M elements, pure streaming).
__global__ __launch_bounds__(256)
void cvt_x_kernel(const float* __restrict__ x, _Float16* __restrict__ xo)
{
    const size_t i = ((size_t)blockIdx.x * 256 + threadIdx.x) * 8;
    const float4* p = (const float4*)(x + i);
    float4 a = p[0], b = p[1];
    v8h h;
    h[0] = (_Float16)a.x; h[1] = (_Float16)a.y; h[2] = (_Float16)a.z; h[3] = (_Float16)a.w;
    h[4] = (_Float16)b.x; h[5] = (_Float16)b.y; h[6] = (_Float16)b.z; h[7] = (_Float16)b.w;
    *(v8h*)(xo + i) = h;
}

// ---------------------------------------------------------------------------
// WMMA 16-bit A/B fragment load per ISA layout:
// lanes 0-15: K{0-7} then K{16-23}; lanes 16-31: K{8-15} then K{24-31}
__device__ __forceinline__ v16h load_frag(const _Float16* base, int row, int lane) {
    const _Float16* p = base + row * BKP + ((lane & 16) ? 8 : 0);
    v8h lo = *(const v8h*)(p);
    v8h hi = *(const v8h*)(p + 16);
    return __builtin_shufflevector(lo, hi, 0,1,2,3,4,5,6,7,8,9,10,11,12,13,14,15);
}

// ---------------------------------------------------------------------------
// Main GEMM, all-f16 operands pre-staged in workspace. Hot loop is
// v_wmma_f32_16x16x32_f16 + ds_load_b128 fragment reads; tile staging uses
// GLOBAL_LOAD_ASYNC_TO_LDS_B128 (ASYNCcnt) when the toolchain provides it.
__global__ __launch_bounds__(256)
void gemm_f16_kernel(const _Float16* __restrict__ A,    // [M][K] f16
                     const _Float16* __restrict__ Bw,   // [N][K] f16
                     const float* __restrict__ bias,
                     float* __restrict__ out)
{
    __shared__ __align__(16) _Float16 As[2][BM * BKP];
    __shared__ __align__(16) _Float16 Bs[2][BN * BKP];

    const int tid  = threadIdx.x;
    const int lane = tid & 31;
    const int wave = tid >> 5;
    const int wm   = wave >> 1;     // 0..3 (64 M-rows each)
    const int wn   = wave & 1;      // 0..1 (64 N-cols each)

    const int m0 = blockIdx.y * BM;
    const int n0 = blockIdx.x * BN;

    const _Float16* gA_row = A + (size_t)(m0 + tid) * K_TOTAL;          // 1 thread / row
    const int rB = tid >> 1;
    const int hB = tid & 1;
    const _Float16* gB_row = Bw + (size_t)(n0 + rB) * K_TOTAL + hB * 16; // pair / row

#if USE_ASYNC_LDS
    // direct global -> LDS DMA, no VGPR round-trip
    auto stage_async = [&](int kt, int buf) {
        const _Float16* ga = gA_row + (size_t)kt * BK;
        _Float16* la = &As[buf][tid * BKP];
        #pragma unroll
        for (int i = 0; i < 4; ++i)
            async_copy_b128(ga + 8 * i, la + 8 * i);
        const _Float16* gb = gB_row + (size_t)kt * BK;
        _Float16* lb = &Bs[buf][rB * BKP + hB * 16];
        async_copy_b128(gb,     lb);
        async_copy_b128(gb + 8, lb + 8);
    };
#else
    v8h aReg[4], bReg[2];
    auto stage_load = [&](int kt) {
        const v8h* ga = (const v8h*)(gA_row + (size_t)kt * BK);
        #pragma unroll
        for (int i = 0; i < 4; ++i) aReg[i] = ga[i];
        const v8h* gb = (const v8h*)(gB_row + (size_t)kt * BK);
        bReg[0] = gb[0];
        bReg[1] = gb[1];
    };
    auto stage_store = [&](int buf) {
        _Float16* a = &As[buf][tid * BKP];
        #pragma unroll
        for (int i = 0; i < 4; ++i) *(v8h*)(a + 8 * i) = aReg[i];
        _Float16* b = &Bs[buf][rB * BKP + hB * 16];
        *(v8h*)(b)     = bReg[0];
        *(v8h*)(b + 8) = bReg[1];
    };
#endif

    v8f acc[4][4];
    {
        v8f z = {0.f,0.f,0.f,0.f,0.f,0.f,0.f,0.f};
        #pragma unroll
        for (int i = 0; i < 4; ++i)
            #pragma unroll
            for (int j = 0; j < 4; ++j) acc[i][j] = z;
    }

#if USE_ASYNC_LDS
    stage_async(0, 0);
    wait_async_zero();
    __syncthreads();
#else
    stage_load(0);
    stage_store(0);
    __syncthreads();
#endif

    for (int kt = 0; kt < NKT; ++kt) {
        const int cur = kt & 1;

        if (kt + 2 < NKT)
            __builtin_prefetch(gA_row + (size_t)(kt + 2) * BK, 0, 1);  // global_prefetch

#if USE_ASYNC_LDS
        if (kt + 1 < NKT) stage_async(kt + 1, cur ^ 1);
#else
        if (kt + 1 < NKT) stage_load(kt + 1);
#endif

        v16h afrag[4], bfrag[4];
        #pragma unroll
        for (int mf = 0; mf < 4; ++mf)
            afrag[mf] = load_frag(As[cur], wm * 64 + mf * 16 + (lane & 15), lane);
        #pragma unroll
        for (int nf = 0; nf < 4; ++nf)
            bfrag[nf] = load_frag(Bs[cur], wn * 64 + nf * 16 + (lane & 15), lane);

        #pragma unroll
        for (int mf = 0; mf < 4; ++mf)
            #pragma unroll
            for (int nf = 0; nf < 4; ++nf)
                acc[mf][nf] = __builtin_amdgcn_wmma_f32_16x16x32_f16(
                    false, afrag[mf], false, bfrag[nf],
                    (short)0, acc[mf][nf], false, false);

#if USE_ASYNC_LDS
        if (kt + 1 < NKT) wait_async_zero();   // next tile landed in LDS
#else
        if (kt + 1 < NKT) stage_store(cur ^ 1);
#endif
        __syncthreads();
    }

    const int col_base = n0 + wn * 64;
    #pragma unroll
    for (int nf = 0; nf < 4; ++nf) {
        const int col = col_base + nf * 16 + (lane & 15);
        const float bv = bias[col];
        #pragma unroll
        for (int mf = 0; mf < 4; ++mf) {
            const int row0 = m0 + wm * 64 + mf * 16 + ((lane >> 4) << 3);
            float* o = out + (size_t)row0 * N_TOTAL + col;
            #pragma unroll
            for (int j = 0; j < 8; ++j)
                o[(size_t)j * N_TOTAL] = acc[mf][nf][j] + bv;
        }
    }
}

// ---------------------------------------------------------------------------
// Fallback: fully fused kernel (used when the workspace is too small).
__global__ __launch_bounds__(256)
void qlinear_fused_kernel(const float* __restrict__ x,
                          const int* __restrict__ wp,
                          const _Float16* __restrict__ wscale,
                          const unsigned char* __restrict__ wmask,
                          const float* __restrict__ bias,
                          float* __restrict__ out)
{
    __shared__ __align__(16) _Float16 As[2][BM * BKP];
    __shared__ __align__(16) _Float16 Bs[2][BN * BKP];

    const int tid  = threadIdx.x;
    const int lane = tid & 31;
    const int wave = tid >> 5;
    const int wm   = wave >> 1;
    const int wn   = wave & 1;

    const int m0 = blockIdx.y * BM;
    const int n0 = blockIdx.x * BN;

    const float* gA_row = x + (size_t)(m0 + tid) * K_TOTAL;
    const int rB = tid >> 1;
    const int hB = tid & 1;
    const size_t nb_base = (size_t)(n0 + rB) * KBLOCKS;

    float4   aReg[8];
    int      bReg[8];
    _Float16 sH = (_Float16)0.0f;

    auto stage_load = [&](int kt) {
        const float4* gA = (const float4*)(gA_row + (size_t)kt * BK);
        #pragma unroll
        for (int i = 0; i < 8; ++i) aReg[i] = gA[i];

        const size_t nb = nb_base + kt;
        const int4* gB = (const int4*)(wp + nb * (GROUP / 2) + hB * 8);
        int4 b0 = gB[0], b1 = gB[1];
        bReg[0] = b0.x; bReg[1] = b0.y; bReg[2] = b0.z; bReg[3] = b0.w;
        bReg[4] = b1.x; bReg[5] = b1.y; bReg[6] = b1.z; bReg[7] = b1.w;
        sH = eff_scale(wscale[nb], wmask[nb]);
    };

    auto stage_store = [&](int buf) {
        _Float16* a = &As[buf][tid * BKP];
        #pragma unroll
        for (int i = 0; i < 4; ++i) {
            v8h p;
            p[0] = (_Float16)aReg[2*i].x;   p[1] = (_Float16)aReg[2*i].y;
            p[2] = (_Float16)aReg[2*i].z;   p[3] = (_Float16)aReg[2*i].w;
            p[4] = (_Float16)aReg[2*i+1].x; p[5] = (_Float16)aReg[2*i+1].y;
            p[6] = (_Float16)aReg[2*i+1].z; p[7] = (_Float16)aReg[2*i+1].w;
            *(v8h*)(a + 8 * i) = p;
        }
        _Float16* b = &Bs[buf][rB * BKP + hB * 16];
        uint4 o0, o1;
        o0.x = dq_pair(bReg[0], sH); o0.y = dq_pair(bReg[1], sH);
        o0.z = dq_pair(bReg[2], sH); o0.w = dq_pair(bReg[3], sH);
        o1.x = dq_pair(bReg[4], sH); o1.y = dq_pair(bReg[5], sH);
        o1.z = dq_pair(bReg[6], sH); o1.w = dq_pair(bReg[7], sH);
        *(uint4*)(b)     = o0;
        *(uint4*)(b + 8) = o1;
    };

    v8f acc[4][4];
    {
        v8f z = {0.f,0.f,0.f,0.f,0.f,0.f,0.f,0.f};
        #pragma unroll
        for (int i = 0; i < 4; ++i)
            #pragma unroll
            for (int j = 0; j < 4; ++j) acc[i][j] = z;
    }

    stage_load(0);
    stage_store(0);
    __syncthreads();

    for (int kt = 0; kt < NKT; ++kt) {
        const int cur = kt & 1;

        if (kt + 2 < NKT)
            __builtin_prefetch(gA_row + (size_t)(kt + 2) * BK, 0, 1);
        if (kt + 1 < NKT) stage_load(kt + 1);

        v16h afrag[4], bfrag[4];
        #pragma unroll
        for (int mf = 0; mf < 4; ++mf)
            afrag[mf] = load_frag(As[cur], wm * 64 + mf * 16 + (lane & 15), lane);
        #pragma unroll
        for (int nf = 0; nf < 4; ++nf)
            bfrag[nf] = load_frag(Bs[cur], wn * 64 + nf * 16 + (lane & 15), lane);

        #pragma unroll
        for (int mf = 0; mf < 4; ++mf)
            #pragma unroll
            for (int nf = 0; nf < 4; ++nf)
                acc[mf][nf] = __builtin_amdgcn_wmma_f32_16x16x32_f16(
                    false, afrag[mf], false, bfrag[nf],
                    (short)0, acc[mf][nf], false, false);

        if (kt + 1 < NKT) stage_store(cur ^ 1);
        __syncthreads();
    }

    const int col_base = n0 + wn * 64;
    #pragma unroll
    for (int nf = 0; nf < 4; ++nf) {
        const int col = col_base + nf * 16 + (lane & 15);
        const float bv = bias[col];
        #pragma unroll
        for (int mf = 0; mf < 4; ++mf) {
            const int row0 = m0 + wm * 64 + mf * 16 + ((lane >> 4) << 3);
            float* o = out + (size_t)row0 * N_TOTAL + col;
            #pragma unroll
            for (int j = 0; j < 8; ++j)
                o[(size_t)j * N_TOTAL] = acc[mf][nf][j] + bv;
        }
    }
}

// ---------------------------------------------------------------------------
extern "C" void kernel_launch(void* const* d_in, const int* in_sizes, int n_in,
                              void* d_out, int out_size, void* d_ws, size_t ws_size,
                              hipStream_t stream) {
    const float*         x     = (const float*)d_in[0];
    const int*           wp    = (const int*)d_in[1];
    const _Float16*      ws    = (const _Float16*)d_in[2];
    const unsigned char* wmask = (const unsigned char*)d_in[3];  // jax bool = 1 byte
    const float*         bias  = (const float*)d_in[4];
    float*               out   = (float*)d_out;

    const size_t w_bytes  = (size_t)N_TOTAL * K_TOTAL * sizeof(_Float16);  //  32 MB
    const size_t x_bytes  = (size_t)M_TOTAL * K_TOTAL * sizeof(_Float16);  // 128 MB

    dim3 grid(N_TOTAL / BN, M_TOTAL / BM);   // (32, 64)

    if (ws_size >= w_bytes + x_bytes) {
        _Float16* wf = (_Float16*)d_ws;
        _Float16* xf = (_Float16*)((char*)d_ws + w_bytes);

        dequant_w_kernel<<<dim3(NB_TOTAL / 256), dim3(256), 0, stream>>>(wp, ws, wmask, wf);
        cvt_x_kernel<<<dim3((unsigned)((size_t)M_TOTAL * K_TOTAL / 8 / 256)),
                       dim3(256), 0, stream>>>(x, xf);
        gemm_f16_kernel<<<grid, dim3(256), 0, stream>>>(xf, wf, bias, out);
    } else {
        qlinear_fused_kernel<<<grid, dim3(256), 0, stream>>>(x, wp, ws, wmask, bias, out);
    }
}
